// multi_attention_31301721654010
// MI455X (gfx1250) — compile-verified, matrix operand order checked by reading, not run on hardware
//
#include <hip/hip_runtime.h>
#include <hip/hip_bf16.h>

// ---------------------------------------------------------------------------
// Multi-head self-attention (shared QKV projection) for MI455X / gfx1250.
// f32 inputs are pre-converted to f16 once (bandwidth-bound pass), then all
// matmuls run through v_wmma_f32_16x16x32_f16 (f16 in, f32 accumulate) with
// pure f16 b128 fragment loads in the hot loops. Flash-attention style:
// the 1 GB score matrix never exists; softmax runs online in registers.
// global_prefetch_b8 issued for next tiles (L2-resident working set, 192 MB).
// ---------------------------------------------------------------------------

typedef __attribute__((ext_vector_type(16))) _Float16 v16h;
typedef __attribute__((ext_vector_type(8)))  _Float16 v8h;
typedef __attribute__((ext_vector_type(8)))  float    v8f;
typedef __attribute__((ext_vector_type(4)))  float    v4f;

#define BATCH 4
#define SEQ   2048
#define DIM   1024
#define NHEAD 16
#define DHEAD 64
#define MTOT  (BATCH * SEQ)       // 8192 rows

union V16U { v16h v; v8h h[2]; };

// CDNA5 16-bit A/B fragment (16x32 / 32x16), per-lane layout:
//   lanes 0-15 : K = ko+{0..7} in halves 0-7, K = ko+{16..23} in halves 8-15, ko=0
//   lanes16-31 : same with ko=8
// p points at element (row_or_col_for_this_lane, kbase + ko).
__device__ inline v16h frag_from_f16(const _Float16* __restrict__ p) {
  V16U u;
  u.h[0] = *(const v8h*)(p);
  u.h[1] = *(const v8h*)(p + 16);
  return u.v;
}

__device__ inline v8f wmma_f16(v16h a, v16h b, v8f c) {
  // D = A(16x32,f16) * B(32x16,f16) + C(16x16,f32)
  return __builtin_amdgcn_wmma_f32_16x16x32_f16(false, a, false, b,
                                                (short)0, c, false, false);
}

// ---------------------------------------------------------------------------
// Kernel 0: bulk f32 -> f16 conversion (pure bandwidth; runs once per tensor)
// ---------------------------------------------------------------------------
__global__ __launch_bounds__(256) void cvt_f32_f16_kernel(
    const float* __restrict__ src, _Float16* __restrict__ dst, int n8) {
  const int i = blockIdx.x * 256 + threadIdx.x;
  if (i >= n8) return;
  const v4f a = *(const v4f*)(src + (size_t)i * 8);
  const v4f b = *(const v4f*)(src + (size_t)i * 8 + 4);
  v8h o;
#pragma unroll
  for (int j = 0; j < 4; ++j) {
    o[j]     = (_Float16)a[j];
    o[4 + j] = (_Float16)b[j];
  }
  *(v8h*)(dst + (size_t)i * 8) = o;
}

// ---------------------------------------------------------------------------
// Kernel 1: qkv = X @ Wqkv^T   (f16 WMMA, f32 accumulate, f16 outputs)
// Also writes the transposed copy qkvT[b][d][s] used as the V B-fragments.
// Block: 128 threads = 4 waves (2x2), block tile 128x128, wave tile 64x64.
// ---------------------------------------------------------------------------
__global__ __launch_bounds__(128) void qkv_gemm_kernel(
    const _Float16* __restrict__ X,   // [8192,1024] f16
    const _Float16* __restrict__ W,   // [1024,1024] f16 (out,in)
    _Float16* __restrict__ qkv,       // [8192,1024]
    _Float16* __restrict__ qkvT) {    // [4][1024][2048]
  const int lane = threadIdx.x & 31;
  const int wave = threadIdx.x >> 5;
  const int l15  = lane & 15;
  const int ko   = (lane >> 4) << 3;        // 0 or 8
  const int mb   = blockIdx.y * 128 + (wave >> 1) * 64;
  const int nb   = blockIdx.x * 128 + (wave & 1) * 64;

  v8f acc[4][4];
#pragma unroll
  for (int mt = 0; mt < 4; ++mt)
#pragma unroll
    for (int nt = 0; nt < 4; ++nt) acc[mt][nt] = (v8f)(0.0f);

  const _Float16* Ab = X + (size_t)(mb + l15) * DIM + ko;
  const _Float16* Bb = W + (size_t)(nb + l15) * DIM + ko;

  for (int k = 0; k < DIM; k += 32) {
    if (k + 128 < DIM) {
      __builtin_prefetch(Ab + k + 128, 0, 1);
      __builtin_prefetch(Bb + k + 128, 0, 1);
    }
    v16h af[4], bf[4];
#pragma unroll
    for (int t = 0; t < 4; ++t) af[t] = frag_from_f16(Ab + (size_t)t * 16 * DIM + k);
#pragma unroll
    for (int t = 0; t < 4; ++t) bf[t] = frag_from_f16(Bb + (size_t)t * 16 * DIM + k);
#pragma unroll
    for (int mt = 0; mt < 4; ++mt)
#pragma unroll
      for (int nt = 0; nt < 4; ++nt)
        acc[mt][nt] = wmma_f16(af[mt], bf[nt], acc[mt][nt]);
  }

  const int rsel = (lane >> 4) << 3;        // C-layout: lanes 16-31 hold rows +8
#pragma unroll
  for (int mt = 0; mt < 4; ++mt)
#pragma unroll
    for (int nt = 0; nt < 4; ++nt)
#pragma unroll
      for (int j = 0; j < 8; ++j) {
        const int r = mb + mt * 16 + rsel + j;
        const int c = nb + nt * 16 + l15;
        const _Float16 v = (_Float16)acc[mt][nt][j];
        qkv[(size_t)r * DIM + c] = v;
        // qkvT[b][c][s] : b = r>>11, s = r&2047
        qkvT[((size_t)(r >> 11) * DIM + c) * SEQ + (r & 2047)] = v;
      }
}

// ---------------------------------------------------------------------------
// Kernel 2: flash attention, q=k=v=qkv. One block = (b,h, 256 q-rows);
// each of 8 waves owns 32 q-rows, loops keys in tiles of 64.
// ---------------------------------------------------------------------------
__global__ __launch_bounds__(256) void attn_kernel(
    const _Float16* __restrict__ qkv,   // [4,2048,1024]
    const _Float16* __restrict__ qkvT,  // [4][1024][2048]
    _Float16* __restrict__ aout) {      // [4,2048,1024]
  __shared__ _Float16 plds[8][32][72];  // wave-private 32x64 P tile (+pad)

  const int lane = threadIdx.x & 31;
  const int wave = threadIdx.x >> 5;
  const int l15  = lane & 15;
  const int ko   = (lane >> 4) << 3;
  const int rsel = (lane >> 4) << 3;

  const int bh = blockIdx.y;
  const int b  = bh >> 4;
  const int h  = bh & 15;
  const int q0 = blockIdx.x * 256 + wave * 32;

  const _Float16* qk = qkv + (size_t)b * SEQ * DIM + h * DHEAD;          // rows: seq
  const _Float16* vT = qkvT + ((size_t)b * DIM + h * DHEAD) * SEQ;       // rows: dim

  // scale folded into exp2 domain: exp(s/8) = 2^(s * log2(e)/8)
  const float SC2 = 0.18033688011112042f;

  // resident Q fragments: [mt][ks]
  v16h qf[2][2];
#pragma unroll
  for (int mt = 0; mt < 2; ++mt)
#pragma unroll
    for (int ks = 0; ks < 2; ++ks)
      qf[mt][ks] = frag_from_f16(qk + (size_t)(q0 + mt * 16 + l15) * DIM + ks * 32 + ko);

  v8f oacc[2][4];
#pragma unroll
  for (int mt = 0; mt < 2; ++mt)
#pragma unroll
    for (int nt = 0; nt < 4; ++nt) oacc[mt][nt] = (v8f)(0.0f);

  float mrow[2][8], lrow[2][8];
#pragma unroll
  for (int mt = 0; mt < 2; ++mt)
#pragma unroll
    for (int j = 0; j < 8; ++j) { mrow[mt][j] = -INFINITY; lrow[mt][j] = 0.0f; }

  for (int kt = 0; kt < SEQ / 64; ++kt) {
    // prefetch next key/value tiles (2 rows per lane each)
    if (kt + 1 < SEQ / 64) {
      const int krow = (kt + 1) * 64 + lane * 2;
      __builtin_prefetch(qk + (size_t)krow * DIM, 0, 1);
      __builtin_prefetch(qk + (size_t)(krow + 1) * DIM, 0, 1);
      __builtin_prefetch(vT + (size_t)(lane * 2) * SEQ + (kt + 1) * 64, 0, 1);
      __builtin_prefetch(vT + (size_t)(lane * 2 + 1) * SEQ + (kt + 1) * 64, 0, 1);
    }

    // ---- scores S = Q @ K^T ---------------------------------------------
    v8f sc[2][4];
#pragma unroll
    for (int mt = 0; mt < 2; ++mt)
#pragma unroll
      for (int nt = 0; nt < 4; ++nt) sc[mt][nt] = (v8f)(0.0f);

#pragma unroll
    for (int ks = 0; ks < 2; ++ks) {
      v16h kf[4];
#pragma unroll
      for (int nt = 0; nt < 4; ++nt)
        kf[nt] = frag_from_f16(qk + (size_t)(kt * 64 + nt * 16 + l15) * DIM + ks * 32 + ko);
#pragma unroll
      for (int mt = 0; mt < 2; ++mt)
#pragma unroll
        for (int nt = 0; nt < 4; ++nt)
          sc[mt][nt] = wmma_f16(qf[mt][ks], kf[nt], sc[mt][nt]);
    }

    // ---- online softmax (row groups of 16 lanes) -------------------------
#pragma unroll
    for (int mt = 0; mt < 2; ++mt)
#pragma unroll
      for (int nt = 0; nt < 4; ++nt)
#pragma unroll
        for (int j = 0; j < 8; ++j) sc[mt][nt][j] *= SC2;

#pragma unroll
    for (int mt = 0; mt < 2; ++mt)
#pragma unroll
      for (int j = 0; j < 8; ++j) {
        float rm = sc[mt][0][j];
#pragma unroll
        for (int nt = 1; nt < 4; ++nt) rm = fmaxf(rm, sc[mt][nt][j]);
#pragma unroll
        for (int off = 1; off < 16; off <<= 1)
          rm = fmaxf(rm, __shfl_xor(rm, off, 32));
        const float mold = mrow[mt][j];
        const float mnew = fmaxf(mold, rm);
        const float corr = exp2f(mold - mnew);
        mrow[mt][j] = mnew;
        float rs = 0.0f;
#pragma unroll
        for (int nt = 0; nt < 4; ++nt) {
          const float p = exp2f(sc[mt][nt][j] - mnew);
          sc[mt][nt][j] = p;
          rs += p;
        }
#pragma unroll
        for (int off = 1; off < 16; off <<= 1)
          rs += __shfl_xor(rs, off, 32);
        lrow[mt][j] = lrow[mt][j] * corr + rs;
#pragma unroll
        for (int nt = 0; nt < 4; ++nt) oacc[mt][nt][j] *= corr;
      }

    // ---- stage P (f16) in LDS, re-read as A fragments --------------------
#pragma unroll
    for (int mt = 0; mt < 2; ++mt)
#pragma unroll
      for (int nt = 0; nt < 4; ++nt)
#pragma unroll
        for (int j = 0; j < 8; ++j)
          plds[wave][mt * 16 + rsel + j][nt * 16 + l15] = (_Float16)sc[mt][nt][j];
    __syncthreads();

    // ---- O += P @ V ------------------------------------------------------
#pragma unroll
    for (int ks = 0; ks < 2; ++ks) {
      v16h pf[2], vf[4];
#pragma unroll
      for (int mt = 0; mt < 2; ++mt)
        pf[mt] = frag_from_f16(&plds[wave][mt * 16 + l15][ks * 32 + ko]);
#pragma unroll
      for (int nt = 0; nt < 4; ++nt)
        vf[nt] = frag_from_f16(vT + (size_t)(nt * 16 + l15) * SEQ + kt * 64 + ks * 32 + ko);
#pragma unroll
      for (int mt = 0; mt < 2; ++mt)
#pragma unroll
        for (int nt = 0; nt < 4; ++nt)
          oacc[mt][nt] = wmma_f16(pf[mt], vf[nt], oacc[mt][nt]);
    }
    __syncthreads();
  }

  // ---- normalize and store f16 attention output --------------------------
#pragma unroll
  for (int mt = 0; mt < 2; ++mt)
#pragma unroll
    for (int j = 0; j < 8; ++j) {
      const float inv = 1.0f / lrow[mt][j];
      const int r = q0 + mt * 16 + rsel + j;
#pragma unroll
      for (int nt = 0; nt < 4; ++nt)
        aout[((size_t)b * SEQ + r) * DIM + h * DHEAD + nt * 16 + l15] =
            (_Float16)(oacc[mt][nt][j] * inv);
    }
}

// ---------------------------------------------------------------------------
// Kernel 3: out = AttnOut @ Wout^T + bias   (f16 WMMA, f32 bias/store)
// ---------------------------------------------------------------------------
__global__ __launch_bounds__(128) void out_gemm_kernel(
    const _Float16* __restrict__ A,   // [8192,1024] f16
    const _Float16* __restrict__ W,   // [1024,1024] f16 (out,in)
    const float* __restrict__ bias,   // [1024] f32
    float* __restrict__ out) {        // [8192,1024] f32
  const int lane = threadIdx.x & 31;
  const int wave = threadIdx.x >> 5;
  const int l15  = lane & 15;
  const int ko   = (lane >> 4) << 3;
  const int mb   = blockIdx.y * 128 + (wave >> 1) * 64;
  const int nb   = blockIdx.x * 128 + (wave & 1) * 64;

  v8f acc[4][4];
#pragma unroll
  for (int mt = 0; mt < 4; ++mt)
#pragma unroll
    for (int nt = 0; nt < 4; ++nt) acc[mt][nt] = (v8f)(0.0f);

  const _Float16* Ab = A + (size_t)(mb + l15) * DIM + ko;
  const _Float16* Bb = W + (size_t)(nb + l15) * DIM + ko;

  for (int k = 0; k < DIM; k += 32) {
    if (k + 128 < DIM) {
      __builtin_prefetch(Ab + k + 128, 0, 1);
      __builtin_prefetch(Bb + k + 128, 0, 1);
    }
    v16h af[4], bf[4];
#pragma unroll
    for (int t = 0; t < 4; ++t) af[t] = frag_from_f16(Ab + (size_t)t * 16 * DIM + k);
#pragma unroll
    for (int t = 0; t < 4; ++t) bf[t] = frag_from_f16(Bb + (size_t)t * 16 * DIM + k);
#pragma unroll
    for (int mt = 0; mt < 4; ++mt)
#pragma unroll
      for (int nt = 0; nt < 4; ++nt)
        acc[mt][nt] = wmma_f16(af[mt], bf[nt], acc[mt][nt]);
  }

  float bv[4];
#pragma unroll
  for (int nt = 0; nt < 4; ++nt) bv[nt] = bias[nb + nt * 16 + l15];

  const int rsel = (lane >> 4) << 3;
#pragma unroll
  for (int mt = 0; mt < 4; ++mt)
#pragma unroll
    for (int nt = 0; nt < 4; ++nt)
#pragma unroll
      for (int j = 0; j < 8; ++j) {
        const int r = mb + mt * 16 + rsel + j;
        const int c = nb + nt * 16 + l15;
        out[(size_t)r * DIM + c] = acc[mt][nt][j] + bv[nt];
      }
}

// ---------------------------------------------------------------------------
extern "C" void kernel_launch(void* const* d_in, const int* in_sizes, int n_in,
                              void* d_out, int out_size, void* d_ws, size_t ws_size,
                              hipStream_t stream) {
  const float* x     = (const float*)d_in[0];   // [4,2048,1024]
  const float* w_qkv = (const float*)d_in[1];   // [1024,1024]
  const float* w_out = (const float*)d_in[2];   // [1024,1024]
  const float* b_out = (const float*)d_in[3];   // [1024]
  float* out = (float*)d_out;

  // workspace (f16): xh 16MB | wqkvh 2MB | wouth 2MB | qkv 16MB | qkvT 16MB | aout 16MB
  _Float16* xh    = (_Float16*)d_ws;
  _Float16* wqkvh = xh    + (size_t)MTOT * DIM;
  _Float16* wouth = wqkvh + (size_t)DIM * DIM;
  _Float16* qkv   = wouth + (size_t)DIM * DIM;
  _Float16* qkvT  = qkv   + (size_t)MTOT * DIM;
  _Float16* aout  = qkvT  + (size_t)MTOT * DIM;

  // ---- pre-convert inputs to f16 (one pass each) ----
  {
    const int n8x = MTOT * DIM / 8;                    // 1,048,576
    cvt_f32_f16_kernel<<<n8x / 256, 256, 0, stream>>>(x, xh, n8x);
    const int n8w = DIM * DIM / 8;                     // 131,072
    cvt_f32_f16_kernel<<<n8w / 256, 256, 0, stream>>>(w_qkv, wqkvh, n8w);
    cvt_f32_f16_kernel<<<n8w / 256, 256, 0, stream>>>(w_out, wouth, n8w);
  }
  {
    dim3 grid(DIM / 128, MTOT / 128);           // (8, 64)
    qkv_gemm_kernel<<<grid, 128, 0, stream>>>(xh, wqkvh, qkv, qkvT);
  }
  {
    dim3 grid(SEQ / 256, BATCH * NHEAD);        // (8, 64)
    attn_kernel<<<grid, 256, 0, stream>>>(qkv, qkvT, aout);
  }
  {
    dim3 grid(DIM / 128, MTOT / 128);           // (8, 64)
    out_gemm_kernel<<<grid, 128, 0, stream>>>(aout, wouth, b_out, out);
  }
}
